// softassignment__63247688401267
// MI455X (gfx1250) — compile-verified, matrix operand order checked by reading, not run on hardware
//
#include <hip/hip_runtime.h>
#include <math.h>
#include <stdint.h>

// CDNA5 / gfx1250 soft-assignment kernel.
//
// Roofline: 128 MB mandatory HBM traffic (64 MB features in, 64 MB out) at
// 23.3 TB/s ~= 5.5 us floor; ~17 GFLOP of f16 WMMA is negligible against the
// 16x16x32 WMMA rate. Design: stream features ONCE via async global->LDS DMA
// (no VGPR round trip), keep the intranormed f16 codebook LDS-resident
// (128 KB of the 320 KB WGP LDS), fuse softmax in-register on the WMMA C
// layout, and use the scale invariance of the final intranorm to drop the
// softmax denominator and constant shifts.

typedef __attribute__((ext_vector_type(16))) _Float16 v16h;
typedef __attribute__((ext_vector_type(8)))  _Float16 v8h;
typedef __attribute__((ext_vector_type(8)))  float    v8f;

#define FDIM   256   // feature dim = NBOOK * DSUB
#define KCODE  256   // codewords per book
#define NBOOK  16
#define DSUB   16
#define WAVES  8
#define TPB    (WAVES * 32)
#define ROWS_PER_WAVE  16
#define ROWS_PER_BLOCK (WAVES * ROWS_PER_WAVE)

// LDS layout (bytes):
//   zsh  : [KCODE][FDIM] f16                 = 131072 (codebook, block-shared)
//   xstf : [WAVES*16][FDIM] f32              = 131072 (async DMA target)
//   sbufT: [WAVES][KCODE][16 rows] f16       =  65536 (soft weights, transposed)
#define Z_BYTES  (KCODE * FDIM * 2)
#define X_BYTES  (WAVES * ROWS_PER_WAVE * FDIM * 4)
#define S_BYTES  (WAVES * KCODE * DSUB * 2)
#define SMEM_BYTES (Z_BYTES + X_BYTES + S_BYTES)   // 327680 = 320 KB (WGP max)

extern "C" __global__ void __launch_bounds__(TPB, 1)
softassign_wmma(const float* __restrict__ feat,
                const float* __restrict__ Zw,
                const float* __restrict__ alphap,
                float* __restrict__ out)
{
    extern __shared__ char smemc[];
    _Float16 (*zsh)[FDIM]  = reinterpret_cast<_Float16(*)[FDIM]>(smemc);
    float    (*xstf)[FDIM] = reinterpret_cast<float(*)[FDIM]>(smemc + Z_BYTES);
    _Float16 (*sbufT)[DSUB]= reinterpret_cast<_Float16(*)[DSUB]>(smemc + Z_BYTES + X_BYTES);

    const int tid  = threadIdx.x;
    const int lane = tid & 31;
    const int wave = tid >> 5;
    const int Lr   = lane & 15;   // row (A) or column (B/C) index within tile
    const int sub  = lane >> 4;   // half-wave selector in the WMMA layouts

    const float alpha = alphap[0];
    const float cexp  = alpha * 1.44269504088896f;  // alpha * log2(e)

    // ------- Stage 1: async DMA this wave's 16 feature rows into LDS -------
    // GLOBAL_LOAD_ASYNC_TO_LDS_B128 (GVS mode): memory never touches VGPRs.
    // 16 rows x 256 f32 = 16 KB per wave, 512 B per issue, 32 issues.
    {
        const unsigned long long sbase = (unsigned long long)(uintptr_t)
            (feat + (size_t)blockIdx.x * (ROWS_PER_BLOCK * FDIM));
        const unsigned xlds = (unsigned)(uintptr_t)(void*)xstf;
        #pragma unroll
        for (int i = 0; i < 32; ++i) {
            const int idx = lane + i * 32;                     // float4 units
            const unsigned off = (unsigned)(wave * (ROWS_PER_WAVE * FDIM * 4)
                                            + idx * 16);       // bytes
            const unsigned ldsb = xlds + off;                  // per-lane LDS addr
            asm volatile("global_load_async_to_lds_b128 %0, %1, %2"
                         :: "v"(ldsb), "v"(off), "s"(sbase)
                         : "memory");
        }
    }

    // ------- Stage 2: intranorm(Z) -> f16 codebook in LDS (overlaps DMA) ---
    // 4096 sub-vectors of length 16; 16 per thread. Z is 256 KB, L2-resident.
    for (int i = 0; i < 16; ++i) {
        const int g = tid * 16 + i;     // sub-vector id
        const int k = g >> 4;           // codeword row
        const int b = g & 15;           // book
        const float* zr = Zw + (size_t)k * FDIM + b * DSUB;
        float v[DSUB];
        float ss = 0.f;
        #pragma unroll
        for (int d = 0; d < DSUB; ++d) { v[d] = zr[d]; ss += v[d] * v[d]; }
        const float inv = 1.0f / fmaxf(sqrtf(ss), 1e-12f);
        #pragma unroll
        for (int d = 0; d < DSUB; ++d)
            zsh[k][b * DSUB + d] = (_Float16)(v[d] * inv);
    }

    asm volatile("s_wait_asynccnt 0" ::: "memory");   // x tile resident in LDS
    __syncthreads();

    const int row0 = (blockIdx.x * WAVES + wave) * ROWS_PER_WAVE;
    const float* xrow = xstf[wave * ROWS_PER_WAVE + Lr];

    // ------- Stage 3: per-book fused GEMM1 / softmax / GEMM2 ---------------
    for (int b = 0; b < NBOOK; ++b) {
        // A fragment: x_b tile [16 rows x K16] cvt f32->f16, K padded to 32.
        // 16-bit A layout: lane (sub,Lr), VGPR j<4 holds K = sub*8 + 2j (+1);
        // VGPRs 4..7 (K 16..31) are zero.
        v16h a;
        #pragma unroll
        for (int j = 0; j < 4; ++j) {
            const int k0 = sub * 8 + j * 2;
            a[2 * j]     = (_Float16)xrow[b * DSUB + k0];
            a[2 * j + 1] = (_Float16)xrow[b * DSUB + k0 + 1];
            a[2 * j + 8] = (_Float16)0.f;
            a[2 * j + 9] = (_Float16)0.f;
        }

        // GEMM1: 16 codeword tiles; accumulators live in 128 VGPRs.
        v8f acc[16];
        #pragma unroll
        for (int t = 0; t < 16; ++t) {
            v16h bf;                      // B: lane column n = t*16+Lr
            const int n = t * 16 + Lr;
            #pragma unroll
            for (int j = 0; j < 4; ++j) {
                const int k0 = sub * 8 + j * 2;
                bf[2 * j]     = zsh[n][b * DSUB + k0];
                bf[2 * j + 1] = zsh[n][b * DSUB + k0 + 1];
                bf[2 * j + 8] = (_Float16)0.f;
                bf[2 * j + 9] = (_Float16)0.f;
            }
            v8f cz = {};
            acc[t] = __builtin_amdgcn_wmma_f32_16x16x32_f16(
                false, a, false, bf, (short)0, cz, false, false);
        }

        // Row max over 256 codewords. C layout: element (t, v, lane) is
        // row = v + 8*sub, col = t*16 + (lane&15): a butterfly over the
        // 16-lane half covers all columns of a row.
        float m[8];
        #pragma unroll
        for (int v = 0; v < 8; ++v) {
            float mm = acc[0][v];
            #pragma unroll
            for (int t = 1; t < 16; ++t) mm = fmaxf(mm, acc[t][v]);
            mm = fmaxf(mm, __shfl_xor(mm, 1));
            mm = fmaxf(mm, __shfl_xor(mm, 2));
            mm = fmaxf(mm, __shfl_xor(mm, 4));
            mm = fmaxf(mm, __shfl_xor(mm, 8));
            m[v] = mm;
        }

        // e = exp(alpha*(s - rowmax)) in (0,1] -> f16-safe. Transposed
        // sbufT[k][row] layout: the 8 rows per (tile,lane) are contiguous,
        // so each tile spills with ONE ds_store_b128 per lane.
        #pragma unroll
        for (int t = 0; t < 16; ++t) {
            v8h ev;
            #pragma unroll
            for (int v = 0; v < 8; ++v)
                ev[v] = (_Float16)exp2f(cexp * (acc[t][v] - m[v]));
            *reinterpret_cast<v8h*>(
                &sbufT[wave * KCODE + t * 16 + Lr][8 * sub]) = ev;
        }
        __syncthreads();

        // GEMM2: desc[16x16] = soft[16x256] * z_b[256x16], K=256 in 8 chunks
        // of 32. A gathers sbufT columns; B gathers z rows.
        v8f dacc = {};
        #pragma unroll
        for (int kc = 0; kc < 8; ++kc) {
            v16h sa, zb;
            #pragma unroll
            for (int j = 0; j < 8; ++j) {
                const int k0 = kc * 32 + (j >> 2) * 16 + sub * 8 + (j & 3) * 2;
                sa[2 * j]     = sbufT[wave * KCODE + k0][Lr];
                sa[2 * j + 1] = sbufT[wave * KCODE + k0 + 1][Lr];
                zb[2 * j]     = zsh[k0][b * DSUB + Lr];
                zb[2 * j + 1] = zsh[k0 + 1][b * DSUB + Lr];
            }
            dacc = __builtin_amdgcn_wmma_f32_16x16x32_f16(
                false, sa, false, zb, (short)0, dacc, false, false);
        }

        // Intranorm on the C layout: sum of squares over the 16 d-columns of
        // a row = butterfly add over the 16-lane half; then scaled store.
        #pragma unroll
        for (int v = 0; v < 8; ++v) {
            const float c = dacc[v];
            float ss = c * c;
            ss += __shfl_xor(ss, 1);
            ss += __shfl_xor(ss, 2);
            ss += __shfl_xor(ss, 4);
            ss += __shfl_xor(ss, 8);
            const float scale = 1.0f / fmaxf(sqrtf(ss), 1e-12f);
            const int r = v + 8 * sub;
            out[(size_t)(row0 + r) * FDIM + b * DSUB + Lr] = c * scale;
        }
        __syncthreads();   // sbufT reused next book
    }
}

extern "C" void kernel_launch(void* const* d_in, const int* in_sizes, int n_in,
                              void* d_out, int out_size, void* d_ws, size_t ws_size,
                              hipStream_t stream) {
    (void)n_in; (void)d_ws; (void)ws_size; (void)out_size;
    const float* feat   = (const float*)d_in[0];   // [N, 256] f32
    const float* Zw     = (const float*)d_in[1];   // [256, 256] f32
    const float* alphap = (const float*)d_in[3];   // scalar f32 on device
    float* out = (float*)d_out;                    // [N, 256] f32

    const int nrows = in_sizes[0] / FDIM;          // 65536
    const int grid  = nrows / ROWS_PER_BLOCK;      // 512

    hipFuncSetAttribute((const void*)softassign_wmma,
                        hipFuncAttributeMaxDynamicSharedMemorySize, SMEM_BYTES);
    hipLaunchKernelGGL(softassign_wmma, dim3(grid), dim3(TPB), SMEM_BYTES, stream,
                       feat, Zw, alphap, out);
}
